// GATModel_22978075033836
// MI455X (gfx1250) — matrix operand order repaired and checked by reading, not compiled, not run on hardware
//
#include <hip/hip_runtime.h>
#include <hip/hip_bf16.h>

#define B_   32
#define L_   256
#define S_   255
#define N_   511
#define H_   768
#define HH_  384
#define M_   16352   /* B_*N_ rows */
#define NB_  16352   /* N_*B_ rows (seq-major) */

typedef __attribute__((ext_vector_type(16))) _Float16 v16h;
typedef __attribute__((ext_vector_type(8)))  float    v8f;

// ---------------- WMMA fragment helpers ----------------
// A fragment (16x32 f16, M x K): lane = half*16 + m, holds K = half*16 + {0..15} contiguous
static __device__ __forceinline__ v16h ldfragA(const _Float16* p) {
  v16h r;
#pragma unroll
  for (int i = 0; i < 16; i++) r[i] = p[i];
  return r;
}
// B fragment (32x16 f16, K x N), column-contiguous storage: lane = half*16 + n,
// holds K = half*16 + {0..15} contiguous
static __device__ __forceinline__ v16h ldfragB(const _Float16* p) {
  v16h r;
#pragma unroll
  for (int i = 0; i < 16; i++) r[i] = p[i];
  return r;
}

// async global -> LDS copy of 16 bytes (per-lane), tracked by ASYNCcnt (CDNA5)
static __device__ __forceinline__ void async_ld_b128(const _Float16* g, const _Float16* l) {
  unsigned lo = (unsigned)(unsigned long long)(const void*)l;      // LDS byte address
  unsigned long long ga = (unsigned long long)(const void*)g;      // 64-bit global address
  asm volatile("global_load_async_to_lds_b128 %0, %1, off" :: "v"(lo), "v"(ga) : "memory");
}

// ---------------- f16 WMMA GEMM with double-buffered async-LDS B tiles ----------------
// C[M,Nd] = A[M,K] * B[K,Nd]; A row-major [M][K] f16; B stored transposed Bt[Nd][K] f16;
// C row-major [M][Nd] f16. Block = 256 thr = 8 waves; block covers 128 rows x 64-col strip.
// Per K-chunk (32), the 64x32 Bt tile (4 KB) is streamed to LDS via async loads (one
// b128 per thread) overlapped with WMMA on the previous chunk. Steady-state loop is
// branch-free; the final chunk is peeled into an epilogue.
__global__ __launch_bounds__(256) void gemm_kernel(const _Float16* __restrict__ A,
                                                   const _Float16* __restrict__ Bt,
                                                   _Float16* __restrict__ C,
                                                   int Mdim, int K, int Nd) {
  __shared__ _Float16 btile[2][64 * 32];

  const int lane = threadIdx.x & 31, wv = threadIdx.x >> 5;
  const int mt = blockIdx.y * 8 + wv;
  const bool mvalid = (mt * 16 < Mdim);
  const int half = lane >> 4, ln = lane & 15;
  const int n0 = blockIdx.x * 64;
  const int row = mvalid ? (mt * 16 + ln) : ln;   // clamp so loads stay in-bounds

  // B staging map: thread -> (col tc, k-offset tk in halves); 64 rows x 64B, 16B per thread
  const int tc = threadIdx.x >> 2;
  const int tk = (threadIdx.x & 3) * 8;
  const _Float16* bsrc = Bt + (size_t)(n0 + tc) * K + tk;

  // prologue: stage chunk 0
  async_ld_b128(bsrc, &btile[0][tc * 32 + tk]);

  v8f acc[4] = {};
  const _Float16* arow = A + (size_t)row * K + half * 16;
  const int kLast = K - 32;   // K is a multiple of 32 and >= 64 here

  for (int k0 = 0; k0 < kLast; k0 += 32) {
    const int p = (k0 >> 5) & 1;
    // prefetch next chunk into the other buffer (unconditional in steady state)
    async_ld_b128(bsrc + (k0 + 32), &btile[p ^ 1][tc * 32 + tk]);

    v16h a = ldfragA(arow + k0);                              // A global loads in flight
    asm volatile("s_wait_asynccnt 0x1" ::: "memory");         // oldest chunk landed
    __syncthreads();                                          // visible to all waves

    v16h b0 = ldfragB(&btile[p][(0 * 16 + ln) * 32 + half * 16]);
    v16h b1 = ldfragB(&btile[p][(1 * 16 + ln) * 32 + half * 16]);
    v16h b2 = ldfragB(&btile[p][(2 * 16 + ln) * 32 + half * 16]);
    v16h b3 = ldfragB(&btile[p][(3 * 16 + ln) * 32 + half * 16]);
    acc[0] = __builtin_amdgcn_wmma_f32_16x16x32_f16(false, a, false, b0, (short)0, acc[0], false, false);
    acc[1] = __builtin_amdgcn_wmma_f32_16x16x32_f16(false, a, false, b1, (short)0, acc[1], false, false);
    acc[2] = __builtin_amdgcn_wmma_f32_16x16x32_f16(false, a, false, b2, (short)0, acc[2], false, false);
    acc[3] = __builtin_amdgcn_wmma_f32_16x16x32_f16(false, a, false, b3, (short)0, acc[3], false, false);
    __syncthreads();                                          // btile[p] free for refill
  }

  { // epilogue: last chunk, nothing in flight behind it
    const int p = (kLast >> 5) & 1;
    v16h a = ldfragA(arow + kLast);
    asm volatile("s_wait_asynccnt 0x0" ::: "memory");
    __syncthreads();
    v16h b0 = ldfragB(&btile[p][(0 * 16 + ln) * 32 + half * 16]);
    v16h b1 = ldfragB(&btile[p][(1 * 16 + ln) * 32 + half * 16]);
    v16h b2 = ldfragB(&btile[p][(2 * 16 + ln) * 32 + half * 16]);
    v16h b3 = ldfragB(&btile[p][(3 * 16 + ln) * 32 + half * 16]);
    acc[0] = __builtin_amdgcn_wmma_f32_16x16x32_f16(false, a, false, b0, (short)0, acc[0], false, false);
    acc[1] = __builtin_amdgcn_wmma_f32_16x16x32_f16(false, a, false, b1, (short)0, acc[1], false, false);
    acc[2] = __builtin_amdgcn_wmma_f32_16x16x32_f16(false, a, false, b2, (short)0, acc[2], false, false);
    acc[3] = __builtin_amdgcn_wmma_f32_16x16x32_f16(false, a, false, b3, (short)0, acc[3], false, false);
  }

  if (!mvalid) return;
#pragma unroll
  for (int nt = 0; nt < 4; nt++) {
    const int col = n0 + nt * 16 + ln;
#pragma unroll
    for (int r = 0; r < 8; r++) {
      const int orow = mt * 16 + r + half * 8;
      C[(size_t)orow * Nd + col] = (_Float16)acc[nt][r];
    }
  }
}

// ---------------- utility kernels ----------------
__global__ void fill_zero_u32(unsigned* p, int n) {
  int i = blockIdx.x * 256 + threadIdx.x;
  if (i < n) p[i] = 0u;
}

__global__ void cvt_copy_f16(const float* __restrict__ in, _Float16* __restrict__ out, int n) {
  int i = blockIdx.x * 256 + threadIdx.x;
  if (i < n) out[i] = (_Float16)in[i];
}

// out[n*K + k] = in[k*Nd + n]   (make weight column-contiguous)
__global__ void cvt_transpose_f16(const float* __restrict__ in, _Float16* __restrict__ out,
                                  int K, int Nd) {
  int t = blockIdx.x * 256 + threadIdx.x;
  if (t >= K * Nd) return;
  int n = t / K, k = t - n * K;
  out[t] = (_Float16)in[(size_t)k * Nd + n];
}

// node features: [2][B][N][H] f16 (word rows from hidden states, span rows from depend_emb gather)
__global__ void nodes_kernel(const float* __restrict__ prem_h, const float* __restrict__ hypo_h,
                             const int* __restrict__ prem_sp, const int* __restrict__ hypo_sp,
                             const float* __restrict__ dep, _Float16* __restrict__ nodes) {
  int id = blockIdx.x;                  // 2*B*N blocks
  int br = id / (B_ * N_);
  int rem = id - br * B_ * N_;
  int b = rem / N_, n = rem - b * N_;
  const float* hs = br ? hypo_h : prem_h;
  const int* sp = br ? hypo_sp : prem_sp;
  const float* src;
  if (n < L_) src = hs + ((size_t)b * L_ + n) * H_;
  else {
    int dd = sp[((size_t)b * S_ + (n - L_)) * 3 + 2];
    src = dep + (size_t)dd * H_;
  }
  _Float16* dst = nodes + ((size_t)br * M_ + (size_t)b * N_ + n) * H_;
  for (int k = threadIdx.x; k < H_; k += 256) dst[k] = (_Float16)src[k];
}

// adjacency bitmask: mask[2][B][N][16] u32
__global__ void mask_kernel(const int* __restrict__ prem_sp, const int* __restrict__ hypo_sp,
                            unsigned* __restrict__ mask) {
  int t = blockIdx.x * 256 + threadIdx.x;
  if (t >= 2 * B_ * S_) return;
  int br = t / (B_ * S_);
  int rem = t - br * B_ * S_;
  int b = rem / S_, s = rem - b * S_;
  const int* sp = br ? hypo_sp : prem_sp;
  int w0 = sp[((size_t)b * S_ + s) * 3 + 0];
  int w1 = sp[((size_t)b * S_ + s) * 3 + 1];
  int d = L_ + s;
  unsigned* mb = mask + (size_t)br * B_ * N_ * 16;
  atomicOr(&mb[((size_t)b * N_ + w0) * 16 + (d >> 5)], 1u << (d & 31));
  atomicOr(&mb[((size_t)b * N_ + d) * 16 + (w0 >> 5)], 1u << (w0 & 31));
  atomicOr(&mb[((size_t)b * N_ + w1) * 16 + (d >> 5)], 1u << (d & 31));
  atomicOr(&mb[((size_t)b * N_ + d) * 16 + (w1 >> 5)], 1u << (w1 & 31));
}

// out[r] = sum_k Z[r*768+k] * avec[k]
__global__ void rowdot_kernel(const _Float16* __restrict__ Z, const float* __restrict__ avec,
                              float* __restrict__ out, int rows) {
  int wv = threadIdx.x >> 5, lane = threadIdx.x & 31;
  int r = blockIdx.x * 4 + wv;
  if (r >= rows) return;
  const _Float16* z = Z + (size_t)r * H_;
  float s = 0.f;
  for (int k = lane; k < H_; k += 32) s += (float)z[k] * avec[k];
#pragma unroll
  for (int o = 16; o > 0; o >>= 1) s += __shfl_xor(s, o);
  if (lane == 0) out[r] = s;
}

// sparse GAT attention layer 1 (per branch): h1cat[b*N+i][head*768 + :] = elu(sum_j al_ij z_j)
__global__ void attn1_kernel(const unsigned* __restrict__ mask, const _Float16* __restrict__ Z,
                             const float* __restrict__ el, const float* __restrict__ er,
                             _Float16* __restrict__ h1cat) {
  const int ROWS = 2 * B_ * N_;
  int wv = threadIdx.x >> 5, lane = threadIdx.x & 31;
  int rid = blockIdx.x * 4 + wv;
  __shared__ int s_js[4][256];
  __shared__ float s_es[4][256];
  __shared__ int s_cnt[4];
  bool valid = rid < ROWS;
  int head = 0, b = 0, i = 0;
  if (lane == 0) s_cnt[wv] = 0;
  __syncthreads();
  if (valid) {
    head = rid / (B_ * N_);
    int rem = rid - head * B_ * N_;
    b = rem / N_;
    i = rem - b * N_;
    float er_i = er[(size_t)head * M_ + b * N_ + i];
    if (lane < 16) {
      unsigned bits = mask[((size_t)b * N_ + i) * 16 + lane];
      while (bits) {
        int bit = __ffs(bits) - 1;
        bits &= bits - 1;
        int j = lane * 32 + bit;
        float e = er_i + el[(size_t)head * M_ + b * N_ + j];
        e = e < 0.f ? 0.2f * e : e;   // leaky_relu(0.2)
        int idx = atomicAdd(&s_cnt[wv], 1);
        s_js[wv][idx] = j;
        s_es[wv][idx] = e;
      }
    }
  }
  __syncthreads();
  int cnt = valid ? s_cnt[wv] : 0;
  float mx = -1e30f;
  for (int n = lane; n < cnt; n += 32) mx = fmaxf(mx, s_es[wv][n]);
#pragma unroll
  for (int o = 16; o > 0; o >>= 1) mx = fmaxf(mx, __shfl_xor(mx, o));
  float sm = 0.f;
  for (int n = lane; n < cnt; n += 32) sm += expf(s_es[wv][n] - mx);
#pragma unroll
  for (int o = 16; o > 0; o >>= 1) sm += __shfl_xor(sm, o);
  float inv = cnt > 0 ? 1.f / sm : 0.f;

  float acc[24];
#pragma unroll
  for (int k = 0; k < 24; k++) acc[k] = 0.f;
  for (int n = 0; n < cnt; n++) {
    float al = expf(s_es[wv][n] - mx) * inv;
    const _Float16* zr = Z + ((size_t)head * M_ + (size_t)b * N_ + s_js[wv][n]) * H_;
#pragma unroll
    for (int k = 0; k < 24; k++) acc[k] += al * (float)zr[lane + 32 * k];
  }
  if (valid) {
    _Float16* out = h1cat + ((size_t)b * N_ + i) * (2 * H_) + (size_t)head * H_;
#pragma unroll
    for (int k = 0; k < 24; k++) {
      float v = acc[k];
      v = v > 0.f ? v : (expf(v) - 1.f);  // elu
      out[lane + 32 * k] = (_Float16)v;
    }
  }
}

// sparse GAT attention layer 2 (per branch); output seq-major gat[(i*B + b)*768 + :] f16
__global__ void attn2_kernel(const unsigned* __restrict__ mask, const _Float16* __restrict__ Z2,
                             const float* __restrict__ el2, const float* __restrict__ er2,
                             _Float16* __restrict__ gat) {
  const int ROWS = B_ * N_;
  int wv = threadIdx.x >> 5, lane = threadIdx.x & 31;
  int rid = blockIdx.x * 4 + wv;
  __shared__ int s_js[4][256];
  __shared__ float s_es[4][256];
  __shared__ int s_cnt[4];
  bool valid = rid < ROWS;
  int b = 0, i = 0;
  if (lane == 0) s_cnt[wv] = 0;
  __syncthreads();
  if (valid) {
    b = rid / N_;
    i = rid - b * N_;
    float er_i = er2[(size_t)b * N_ + i];
    if (lane < 16) {
      unsigned bits = mask[((size_t)b * N_ + i) * 16 + lane];
      while (bits) {
        int bit = __ffs(bits) - 1;
        bits &= bits - 1;
        int j = lane * 32 + bit;
        float e = er_i + el2[(size_t)b * N_ + j];
        e = e < 0.f ? 0.2f * e : e;
        int idx = atomicAdd(&s_cnt[wv], 1);
        s_js[wv][idx] = j;
        s_es[wv][idx] = e;
      }
    }
  }
  __syncthreads();
  int cnt = valid ? s_cnt[wv] : 0;
  float mx = -1e30f;
  for (int n = lane; n < cnt; n += 32) mx = fmaxf(mx, s_es[wv][n]);
#pragma unroll
  for (int o = 16; o > 0; o >>= 1) mx = fmaxf(mx, __shfl_xor(mx, o));
  float sm = 0.f;
  for (int n = lane; n < cnt; n += 32) sm += expf(s_es[wv][n] - mx);
#pragma unroll
  for (int o = 16; o > 0; o >>= 1) sm += __shfl_xor(sm, o);
  float inv = cnt > 0 ? 1.f / sm : 0.f;

  float acc[24];
#pragma unroll
  for (int k = 0; k < 24; k++) acc[k] = 0.f;
  for (int n = 0; n < cnt; n++) {
    float al = expf(s_es[wv][n] - mx) * inv;
    const _Float16* zr = Z2 + ((size_t)b * N_ + s_js[wv][n]) * H_;
#pragma unroll
    for (int k = 0; k < 24; k++) acc[k] += al * (float)zr[lane + 32 * k];
  }
  if (valid) {
    _Float16* out = gat + ((size_t)i * B_ + b) * H_;  // seq-major for LSTM
#pragma unroll
    for (int k = 0; k < 24; k++) out[lane + 32 * k] = (_Float16)acc[k];
  }
}

// ---------------- persistent WMMA LSTM recurrence ----------------
// One block per recurrence (rec: 0=lstm1_f 1=lstm1_b 2=lstm2_f 3=lstm2_b).
// pre[rec][t*32+b][1536] f16 = x_t @ Wih^T (precomputed). WhhT[rec][1536][384] f16.
// h double-buffered f16 in LDS; c lives in registers in WMMA C-layout.
__global__ __launch_bounds__(256) void lstm_kernel(const _Float16* __restrict__ pre,
                                                   const _Float16* __restrict__ WhhT,
                                                   const float* __restrict__ b0,
                                                   const float* __restrict__ b1,
                                                   const float* __restrict__ b2,
                                                   const float* __restrict__ b3,
                                                   float* __restrict__ hout) {
  const int rec = blockIdx.x;
  const float* bias = rec == 0 ? b0 : rec == 1 ? b1 : rec == 2 ? b2 : b3;
  const _Float16* W = WhhT + (size_t)rec * 1536 * HH_;
  const _Float16* P = pre + (size_t)rec * NB_ * 1536;
  const bool bwd = (rec & 1);

  __shared__ _Float16 hbuf[2][32 * HH_];
  const int tid = threadIdx.x, lane = tid & 31, wv = tid >> 5;
  const int half = lane >> 4, ln = lane & 15;

  for (int k = tid; k < 32 * HH_; k += 256) hbuf[0][k] = (_Float16)0.f;
  float c[3][2][8];
#pragma unroll
  for (int q = 0; q < 3; q++)
#pragma unroll
    for (int mt = 0; mt < 2; mt++)
#pragma unroll
      for (int r = 0; r < 8; r++) c[q][mt][r] = 0.f;
  __syncthreads();

  for (int t = 0; t < N_; t++) {
    const int seq = bwd ? (N_ - 1 - t) : t;
    const _Float16* hsrc = hbuf[t & 1];
    _Float16* hdst = hbuf[(t + 1) & 1];

    for (int q = 0; q < 3; q++) {
      const int jb = wv * 3 + q;                 // 24 blocks of 16 hidden cols
      v8f acc[2][4] = {};
      for (int k0 = 0; k0 < HH_; k0 += 32) {
        v16h a0 = ldfragA(hsrc + (size_t)(ln) * HH_ + k0 + half * 16);
        v16h a1 = ldfragA(hsrc + (size_t)(16 + ln) * HH_ + k0 + half * 16);
        v16h bb[4];
#pragma unroll
        for (int g = 0; g < 4; g++)
          bb[g] = ldfragB(W + (size_t)(g * HH_ + jb * 16 + ln) * HH_ + k0 + half * 16);
#pragma unroll
        for (int g = 0; g < 4; g++) {
          acc[0][g] = __builtin_amdgcn_wmma_f32_16x16x32_f16(false, a0, false, bb[g],
                                                             (short)0, acc[0][g], false, false);
          acc[1][g] = __builtin_amdgcn_wmma_f32_16x16x32_f16(false, a1, false, bb[g],
                                                             (short)0, acc[1][g], false, false);
        }
      }
#pragma unroll
      for (int mt = 0; mt < 2; mt++) {
#pragma unroll
        for (int r = 0; r < 8; r++) {
          const int row = mt * 16 + r + half * 8;   // batch index 0..31
          const int j = jb * 16 + ln;               // hidden index 0..383
          const _Float16* pr = P + ((size_t)seq * 32 + row) * 1536;
          float gi = acc[mt][0][r] + (float)pr[0 * HH_ + j] + bias[0 * HH_ + j];
          float gf = acc[mt][1][r] + (float)pr[1 * HH_ + j] + bias[1 * HH_ + j];
          float gg = acc[mt][2][r] + (float)pr[2 * HH_ + j] + bias[2 * HH_ + j];
          float go = acc[mt][3][r] + (float)pr[3 * HH_ + j] + bias[3 * HH_ + j];
          float si = 1.f / (1.f + expf(-gi));
          float sf = 1.f / (1.f + expf(-gf));
          float so = 1.f / (1.f + expf(-go));
          float cc = sf * c[q][mt][r] + si * tanhf(gg);
          c[q][mt][r] = cc;
          hdst[(size_t)row * HH_ + j] = (_Float16)(so * tanhf(cc));
        }
      }
    }
    __syncthreads();
  }
  const _Float16* hf = hbuf[N_ & 1];
  for (int k = tid; k < 32 * HH_; k += 256)
    hout[(size_t)rec * 32 * HH_ + k] = (float)hf[k];
}

// ---------------- bilinear head: out[b,k] = p_h . (W_k h_h) + b_k ----------------
__global__ void bilinear_kernel(const float* __restrict__ hout, const float* __restrict__ bilW,
                                const float* __restrict__ bilb, float* __restrict__ out) {
  const int b = blockIdx.x, k = blockIdx.y;
  __shared__ float sh[768];
  __shared__ float red[256];
  const float* h2 = hout + 2 * 32 * HH_;
  const float* h3 = hout + 3 * 32 * HH_;
  for (int j = threadIdx.x; j < 768; j += 256)
    sh[j] = (j < HH_) ? h2[(size_t)b * HH_ + j] : h3[(size_t)b * HH_ + j - HH_];
  __syncthreads();
  const float* h0 = hout;
  const float* h1 = hout + 32 * HH_;
  const float* Wk = bilW + (size_t)k * 768 * 768;
  float part = 0.f;
  for (int i = threadIdx.x; i < 768; i += 256) {
    float ph = (i < HH_) ? h0[(size_t)b * HH_ + i] : h1[(size_t)b * HH_ + i - HH_];
    float dot = 0.f;
    for (int j = 0; j < 768; j++) dot += Wk[(size_t)i * 768 + j] * sh[j];
    part += ph * dot;
  }
  red[threadIdx.x] = part;
  __syncthreads();
  for (int s = 128; s > 0; s >>= 1) {
    if (threadIdx.x < s) red[threadIdx.x] += red[threadIdx.x + s];
    __syncthreads();
  }
  if (threadIdx.x == 0) out[b * 3 + k] = red[0] + bilb[k];
}

// ---------------- host orchestration ----------------
extern "C" void kernel_launch(void* const* d_in, const int* in_sizes, int n_in,
                              void* d_out, int out_size, void* d_ws, size_t ws_size,
                              hipStream_t stream) {
  const float* prem_h  = (const float*)d_in[0];
  const float* hypo_h  = (const float*)d_in[1];
  const int*   prem_sp = (const int*)d_in[2];
  const int*   hypo_sp = (const int*)d_in[3];
  const float* dep     = (const float*)d_in[5];
  const float* W1f[2]  = {(const float*)d_in[6],  (const float*)d_in[10]};
  const float* a1f[2]  = {(const float*)d_in[7],  (const float*)d_in[11]};
  const float* W2f[2]  = {(const float*)d_in[8],  (const float*)d_in[12]};
  const float* a2f[2]  = {(const float*)d_in[9],  (const float*)d_in[13]};
  const float* bilW    = (const float*)d_in[14];
  const float* bilb    = (const float*)d_in[15];
  const float* Wih[4]  = {(const float*)d_in[16], (const float*)d_in[19],
                          (const float*)d_in[22], (const float*)d_in[25]};
  const float* Whh[4]  = {(const float*)d_in[17], (const float*)d_in[20],
                          (const float*)d_in[23], (const float*)d_in[26]};
  const float* bls[4]  = {(const float*)d_in[18], (const float*)d_in[21],
                          (const float*)d_in[24], (const float*)d_in[27]};

  char* w = (char*)d_ws;
  size_t off = 0;
  auto alloc = [&](size_t bytes) {
    size_t o = off;
    off += (bytes + 255) & ~(size_t)255;
    return o;
  };
  _Float16* nodes = (_Float16*)(w + alloc((size_t)2 * M_ * H_ * 2));
  unsigned* mask  = (unsigned*)(w + alloc((size_t)2 * B_ * N_ * 16 * 4));
  _Float16* W1t   = (_Float16*)(w + alloc((size_t)4 * H_ * H_ * 2));
  _Float16* W2t   = (_Float16*)(w + alloc((size_t)2 * H_ * 1536 * 2));
  _Float16* WihT  = (_Float16*)(w + alloc((size_t)4 * 1536 * H_ * 2));
  _Float16* WhhT  = (_Float16*)(w + alloc((size_t)4 * 1536 * HH_ * 2));
  _Float16* Z     = (_Float16*)(w + alloc((size_t)4 * M_ * H_ * 2));
  float*    el    = (float*)(w + alloc((size_t)4 * M_ * 4));
  float*    er    = (float*)(w + alloc((size_t)4 * M_ * 4));
  _Float16* h1cat = (_Float16*)(w + alloc((size_t)2 * M_ * 1536 * 2));
  _Float16* z2    = (_Float16*)(w + alloc((size_t)2 * M_ * H_ * 2));
  float*    el2   = (float*)(w + alloc((size_t)2 * M_ * 4));
  float*    er2   = (float*)(w + alloc((size_t)2 * M_ * 4));
  _Float16* gat   = (_Float16*)(w + alloc((size_t)2 * NB_ * H_ * 2));
  _Float16* pre   = (_Float16*)(w + alloc((size_t)4 * NB_ * 1536 * 2));
  float*    hout  = (float*)(w + alloc((size_t)4 * 32 * HH_ * 4));
  (void)ws_size; (void)in_sizes; (void)n_in; (void)out_size;

  // 0) zero adjacency bitmask (atomicOr target)
  {
    int nw = 2 * B_ * N_ * 16;
    fill_zero_u32<<<(nw + 255) / 256, 256, 0, stream>>>(mask, nw);
  }
  // 1) node features + adjacency
  nodes_kernel<<<2 * B_ * N_, 256, 0, stream>>>(prem_h, hypo_h, prem_sp, hypo_sp, dep, nodes);
  mask_kernel<<<(2 * B_ * S_ + 255) / 256, 256, 0, stream>>>(prem_sp, hypo_sp, mask);

  // 2) weight conversions (f16, column-contiguous)
  for (int br = 0; br < 2; br++)
    for (int hd = 0; hd < 2; hd++)
      cvt_transpose_f16<<<(H_ * H_ + 255) / 256, 256, 0, stream>>>(
          W1f[br] + (size_t)hd * H_ * H_, W1t + ((size_t)br * 2 + hd) * H_ * H_, H_, H_);
  for (int br = 0; br < 2; br++)
    cvt_transpose_f16<<<(1536 * H_ + 255) / 256, 256, 0, stream>>>(
        W2f[br], W2t + (size_t)br * H_ * 1536, 1536, H_);
  for (int r = 0; r < 4; r++) {
    cvt_copy_f16<<<(1536 * H_ + 255) / 256, 256, 0, stream>>>(Wih[r], WihT + (size_t)r * 1536 * H_,
                                                              1536 * H_);
    cvt_copy_f16<<<(1536 * HH_ + 255) / 256, 256, 0, stream>>>(Whh[r], WhhT + (size_t)r * 1536 * HH_,
                                                               1536 * HH_);
  }

  // 3) GAT layer 1 projections: Z[br,hd] = nodes[br] @ W1[br,hd]   (WMMA, async-LDS B)
  for (int br = 0; br < 2; br++)
    for (int hd = 0; hd < 2; hd++)
      gemm_kernel<<<dim3(H_ / 64, 128), 256, 0, stream>>>(
          nodes + (size_t)br * M_ * H_, W1t + ((size_t)br * 2 + hd) * H_ * H_,
          Z + ((size_t)br * 2 + hd) * M_ * H_, M_, H_, H_);

  // 4) attention logits el/er
  for (int br = 0; br < 2; br++)
    for (int hd = 0; hd < 2; hd++) {
      const _Float16* zp = Z + ((size_t)br * 2 + hd) * M_ * H_;
      rowdot_kernel<<<(M_ + 3) / 4, 128, 0, stream>>>(zp, a1f[br] + (size_t)hd * 1536,
                                                      el + ((size_t)br * 2 + hd) * M_, M_);
      rowdot_kernel<<<(M_ + 3) / 4, 128, 0, stream>>>(zp, a1f[br] + (size_t)hd * 1536 + H_,
                                                      er + ((size_t)br * 2 + hd) * M_, M_);
    }

  // 5) sparse attention layer 1 + ELU -> h1cat
  for (int br = 0; br < 2; br++)
    attn1_kernel<<<(2 * B_ * N_ + 3) / 4, 128, 0, stream>>>(
        mask + (size_t)br * B_ * N_ * 16, Z + (size_t)br * 2 * M_ * H_,
        el + (size_t)br * 2 * M_, er + (size_t)br * 2 * M_, h1cat + (size_t)br * M_ * 1536);

  // 6) GAT layer 2 projection: z2 = h1cat @ W2   (WMMA, K=1536)
  for (int br = 0; br < 2; br++)
    gemm_kernel<<<dim3(H_ / 64, 128), 256, 0, stream>>>(
        h1cat + (size_t)br * M_ * 1536, W2t + (size_t)br * H_ * 1536,
        z2 + (size_t)br * M_ * H_, M_, 1536, H_);

  // 7) layer-2 logits + sparse attention -> gat (seq-major f16)
  for (int br = 0; br < 2; br++) {
    const _Float16* zp = z2 + (size_t)br * M_ * H_;
    rowdot_kernel<<<(M_ + 3) / 4, 128, 0, stream>>>(zp, a2f[br], el2 + (size_t)br * M_, M_);
    rowdot_kernel<<<(M_ + 3) / 4, 128, 0, stream>>>(zp, a2f[br] + H_, er2 + (size_t)br * M_, M_);
    attn2_kernel<<<(B_ * N_ + 3) / 4, 128, 0, stream>>>(
        mask + (size_t)br * B_ * N_ * 16, zp, el2 + (size_t)br * M_, er2 + (size_t)br * M_,
        gat + (size_t)br * NB_ * H_);
  }

  // 8) LSTM input projections: pre[rec] = seq @ Wih[rec]^T   (WMMA, Nd=1536)
  for (int r = 0; r < 4; r++) {
    const _Float16* A = gat + (size_t)(r / 2) * NB_ * H_;  // rec0,1 -> prem; rec2,3 -> hypo
    gemm_kernel<<<dim3(1536 / 64, 128), 256, 0, stream>>>(
        A, WihT + (size_t)r * 1536 * H_, pre + (size_t)r * NB_ * 1536, NB_, H_, 1536);
  }

  // 9) persistent WMMA LSTM recurrences (4 blocks, one per direction/lstm)
  lstm_kernel<<<4, 256, 0, stream>>>(pre, WhhT, bls[0], bls[1], bls[2], bls[3], hout);

  // 10) bilinear head
  bilinear_kernel<<<dim3(B_, 3), 256, 0, stream>>>(hout, bilW, bilb, (float*)d_out);
}
